// QnADownBlock_87282325389576
// MI455X (gfx1250) — matrix operand order repaired and checked by reading, not compiled, not run on hardware
//
#include <hip/hip_runtime.h>
#include <hip/hip_bf16.h>

typedef __attribute__((ext_vector_type(16))) __bf16 v16bf;
typedef __attribute__((ext_vector_type(8)))  __bf16 v8bf;
typedef __attribute__((ext_vector_type(8)))  float  v8f;

#define QNA_B     16
#define QNA_C     256
#define QNA_T     8192
#define QNA_TF    512
#define QNA_HID   256
#define QNA_T2    4096
#define QNA_EPS   1e-6f

#define TILE_T    64      // time columns per workgroup
#define TILE_T2   32      // output columns per workgroup
#define XMP       272     // xm LDS row stride (halves), 544B = 17*32 -> 32B aligned rows
#define KVP       528     // kv LDS row stride (halves), 1056B
#define OPD       272     // o  LDS row stride (halves)

// ---- dynamic LDS layout (bytes) ----
#define OFF_XM    0
#define OFF_KV    (TILE_T * XMP * 2)               // 34816
#define OFF_O     (OFF_KV + TILE_T * KVP * 2)      // 102400
#define OFF_F     (OFF_O + TILE_T2 * OPD * 2)      // 119808
#define OFF_GAMMA (OFF_F)
#define OFF_BETA  (OFF_GAMMA + 256*4)
#define OFF_KVB   (OFF_BETA  + 256*4)
#define OFF_QB    (OFF_KVB   + 512*4)
#define OFF_PJB   (OFF_QB    + 256*4)
#define OFF_RPE   (OFF_PJB   + 256*4)
#define OFF_MU    (OFF_RPE   + 16*4)
#define OFF_RI    (OFF_MU    + 64*4)
#define OFF_RSUM  (OFF_RI    + 64*4)
#define OFF_RSQ   (OFF_RSUM  + 256*4)
#define SMEM_BYTES (OFF_RSQ + 256*4)               // 128576 B

// ---- workspace layout (bytes) ----
#define WS_KVPK   0                                // 256 frags * 512 bf16 = 262144 B
#define WS_PJPK   262144                           // 128 frags * 512 bf16 = 131072 B
#define WS_COND   393216                           // 16*512 f32
#define WS_QB     425984                           // 16*256 f32

// =====================================================================
// Kernel 1: pack kv_w [512x256] and proj_w [256x256] (fp32, row-major)
// into bf16 WMMA A-fragment order: frag = (mtile * 8 + ktile), per lane
// 16 contiguous bf16 in contiguous k order (B reads match):
//   lane<16 : row = mtile*16+lane,    K(i) = i
//   lane>=16: row = mtile*16+lane-16, K(i) = i+16
// =====================================================================
__global__ __launch_bounds__(256) void qna_pack_w(
    const float* __restrict__ kvw, const float* __restrict__ pjw,
    __bf16* __restrict__ kvpk, __bf16* __restrict__ pjpk)
{
    int t = blockIdx.x * blockDim.x + threadIdx.x;
    const float* src;
    __bf16* dst;
    if (t < 256 * 32) {                 // kv_w: 32 mtiles * 8 ktiles
        src = kvw; dst = kvpk;
    } else {
        t -= 256 * 32;
        if (t >= 128 * 32) return;      // proj_w: 16 mtiles * 8 ktiles
        src = pjw; dst = pjpk;
    }
    const int frag = t >> 5;
    const int lane = t & 31;
    const int mt = frag >> 3, kt = frag & 7;
    const int row = mt * 16 + (lane & 15);
    __bf16* d = dst + (size_t)(frag * 32 + lane) * 16;
#pragma unroll
    for (int i = 0; i < 16; ++i) {
        int K = (lane < 16) ? i
                            : i + 16;
        d[i] = (__bf16)src[(size_t)row * QNA_C + kt * 32 + K];
    }
}

// =====================================================================
// Kernel 2: per-batch FiLM cond = silu(temb) @ ss_w.T + ss_b  [B,512]
//           and qb = q + silu(temb) @ qt_w.T + qt_b           [B,256]
// =====================================================================
__global__ __launch_bounds__(256) void qna_cond(
    const float* __restrict__ temb, const float* __restrict__ ss_w,
    const float* __restrict__ ss_b, const float* __restrict__ qt_w,
    const float* __restrict__ qt_b, const float* __restrict__ q,
    float* __restrict__ cond_o, float* __restrict__ qb_o)
{
    __shared__ float s[QNA_TF];
    const int b = blockIdx.x, tid = threadIdx.x;
    for (int i = tid; i < QNA_TF; i += 256) {
        float v = temb[b * QNA_TF + i];
        s[i] = v / (1.f + __expf(-v));         // silu
    }
    __syncthreads();
    for (int r = tid; r < 2 * QNA_C; r += 256) {
        const float* w = ss_w + (size_t)r * QNA_TF;
        float acc = 0.f;
        for (int f = 0; f < QNA_TF; ++f) acc += s[f] * w[f];
        cond_o[b * 512 + r] = acc + ss_b[r];
    }
    {
        const float* w = qt_w + (size_t)tid * QNA_TF;
        float acc = 0.f;
        for (int f = 0; f < QNA_TF; ++f) acc += s[f] * w[f];
        qb_o[b * 256 + tid] = q[tid] + acc + qt_b[tid];
    }
}

// =====================================================================
// Kernel 3 (main): fused LN + FiLM + kv GEMM (WMMA bf16) + K=2 window
// attention + proj GEMM (WMMA bf16).  One WG = one (batch, 64-col tile).
// =====================================================================
__global__ __launch_bounds__(256) void qna_main(
    const float* __restrict__ x,    const float* __restrict__ ln_w,
    const float* __restrict__ ln_b, const float* __restrict__ kv_b,
    const float* __restrict__ rpe,  const float* __restrict__ proj_b,
    const float* __restrict__ cond, const float* __restrict__ qbv,
    const __bf16* __restrict__ kv_pk, const __bf16* __restrict__ pj_pk,
    float* __restrict__ out)
{
    extern __shared__ __align__(32) char smem[];
    __bf16* xm_l  = (__bf16*)(smem + OFF_XM);   // [TILE_T][XMP]  time-major, bf16
    __bf16* kv_l  = (__bf16*)(smem + OFF_KV);   // [TILE_T][KVP]  time-major, bf16
    __bf16* o_l   = (__bf16*)(smem + OFF_O);    // [TILE_T2][OPD] time-major, bf16
    float*  gam_l = (float*)(smem + OFF_GAMMA);
    float*  bet_l = (float*)(smem + OFF_BETA);
    float*  kvb_l = (float*)(smem + OFF_KVB);
    float*  qb_l  = (float*)(smem + OFF_QB);
    float*  pjb_l = (float*)(smem + OFF_PJB);
    float*  rpe_l = (float*)(smem + OFF_RPE);
    float*  mu_l  = (float*)(smem + OFF_MU);
    float*  ri_l  = (float*)(smem + OFF_RI);
    float*  rsum  = (float*)(smem + OFF_RSUM);
    float*  rsq   = (float*)(smem + OFF_RSQ);

    const int b    = blockIdx.x >> 7;           // 128 tiles per batch
    const int tile = blockIdx.x & 127;
    const int t0   = tile * TILE_T;
    const int tid  = threadIdx.x;
    const int lane = tid & 31, wave = tid >> 5;

    // ---- Phase 0: per-channel params (fused LN*FiLM coefficients) ----
    {
        float sc = cond[b * 512 + 256 + tid];   // scale
        float sh = cond[b * 512 + tid];         // shift
        gam_l[tid] = ln_w[tid] * (1.f + sc);
        bet_l[tid] = ln_b[tid] * (1.f + sc) + sh;
        qb_l[tid]  = qbv[b * 256 + tid];
        pjb_l[tid] = proj_b[tid];
        kvb_l[tid]       = kv_b[tid];
        kvb_l[tid + 256] = kv_b[tid + 256];
        if (tid < 16) rpe_l[tid] = rpe[tid];
    }

    // ---- Phase 1: load x (coalesced, 4 threads/column), LN stats ----
    const int col = tid & 63, qq = tid >> 6;    // column in tile, channel quarter
    float xv[64];
    {
        const float* xp = x + ((size_t)b * QNA_C + qq * 64) * QNA_T + t0 + col;
        float s = 0.f, s2 = 0.f;
#pragma unroll
        for (int i = 0; i < 64; ++i) {
            float v = xp[(size_t)i * QNA_T];
            xv[i] = v; s += v; s2 += v * v;
        }
        rsum[tid] = s; rsq[tid] = s2;
    }
    __syncthreads();
    if (tid < 64) {
        float ts = rsum[tid] + rsum[tid + 64] + rsum[tid + 128] + rsum[tid + 192];
        float t2 = rsq[tid]  + rsq[tid + 64]  + rsq[tid + 128]  + rsq[tid + 192];
        float mu  = ts * (1.f / 256.f);
        float var = t2 * (1.f / 256.f) - mu * mu;
        mu_l[tid] = mu;
        ri_l[tid] = rsqrtf(var + QNA_EPS);
    }
    __syncthreads();
    {   // normalize + FiLM, write bf16 time-major (B-operand layout)
        float mu = mu_l[col], ri = ri_l[col];
#pragma unroll
        for (int i = 0; i < 64; i += 8) {
            v8bf pk;
#pragma unroll
            for (int j = 0; j < 8; ++j) {
                int c = qq * 64 + i + j;
                pk[j] = (__bf16)(((xv[i + j] - mu) * ri) * gam_l[c] + bet_l[c]);
            }
            *(v8bf*)&xm_l[col * XMP + qq * 64 + i] = pk;
        }
    }
    __syncthreads();

    // ---- Phase 2: kv GEMM: kv[512,64] = kv_w[512,256] x xm[256,64] ----
    {
        const int nRow  = lane & 15;
        const int kHalf = (lane >> 4) * 16;
        for (int nt = 0; nt < 4; ++nt) {        // 4 N-tiles of 16
            v16bf Bf[8];
#pragma unroll
            for (int kt = 0; kt < 8; ++kt)
                Bf[kt] = *(const v16bf*)&xm_l[(nt * 16 + nRow) * XMP + kt * 32 + kHalf];
#pragma unroll
            for (int m4 = 0; m4 < 4; ++m4) {    // 4 M-tiles per wave
                const int mt = wave * 4 + m4;   // 0..31
                v8f acc = {};
#pragma unroll
                for (int kt = 0; kt < 8; ++kt) {
                    v16bf Af = *(const v16bf*)&kv_pk[(size_t)((mt * 8 + kt) * 32 + lane) * 16];
                    acc = __builtin_amdgcn_wmma_f32_16x16x32_bf16(
                        false, Af, false, Bf[kt], (short)0, acc, false, false);
                }
                const int mrow = mt * 16 + ((lane >> 4) << 3);  // C-layout rows
                const int tcol = nt * 16 + nRow;
                v8bf pk;
#pragma unroll
                for (int j = 0; j < 8; ++j)
                    pk[j] = (__bf16)(acc[j] + kvb_l[mrow + j]);
                *(v8bf*)&kv_l[tcol * KVP + mrow] = pk;
            }
        }
    }
    __syncthreads();

    // ---- Phase 3: K=2 window attention, one (head,t2) per thread ----
    {
        const int h = tid >> 5, t2i = tid & 31;
        const __bf16* k0 = &kv_l[(2 * t2i)     * KVP + h * 32];
        const __bf16* k1 = &kv_l[(2 * t2i + 1) * KVP + h * 32];
        float l0 = 0.f, l1 = 0.f;
#pragma unroll
        for (int d = 0; d < 32; ++d) {
            float qd = qb_l[h * 32 + d];
            l0 += qd * (float)k0[d];
            l1 += qd * (float)k1[d];
        }
        const float rs = 0.17677669529663687f;  // 32^-0.5
        l0 = l0 * rs + rpe_l[h * 2 + 0];
        l1 = l1 * rs + rpe_l[h * 2 + 1];
        float m  = fmaxf(l0, l1);
        float e0 = __expf(l0 - m), e1 = __expf(l1 - m);
        float inv = 1.f / (e0 + e1);
        float w0 = e0 * inv, w1 = e1 * inv;
        const __bf16* v0 = k0 + 256;            // v rows live at channel+256
        const __bf16* v1 = k1 + 256;
#pragma unroll
        for (int d = 0; d < 32; d += 8) {
            v8bf pk;
#pragma unroll
            for (int j = 0; j < 8; ++j)
                pk[j] = (__bf16)(w0 * (float)v0[d + j] + w1 * (float)v1[d + j]);
            *(v8bf*)&o_l[t2i * OPD + h * 32 + d] = pk;
        }
    }
    __syncthreads();

    // ---- Phase 4: proj GEMM: out[256,32] = proj_w[256,256] x o[256,32] ----
    {
        const int nRow  = lane & 15;
        const int kHalf = (lane >> 4) * 16;
        const int t2base = t0 >> 1;             // tile*32
        for (int nt = 0; nt < 2; ++nt) {
            v16bf Bf[8];
#pragma unroll
            for (int kt = 0; kt < 8; ++kt)
                Bf[kt] = *(const v16bf*)&o_l[(nt * 16 + nRow) * OPD + kt * 32 + kHalf];
#pragma unroll
            for (int m2 = 0; m2 < 2; ++m2) {
                const int mt = wave * 2 + m2;   // 0..15
                v8f acc = {};
#pragma unroll
                for (int kt = 0; kt < 8; ++kt) {
                    v16bf Af = *(const v16bf*)&pj_pk[(size_t)((mt * 8 + kt) * 32 + lane) * 16];
                    acc = __builtin_amdgcn_wmma_f32_16x16x32_bf16(
                        false, Af, false, Bf[kt], (short)0, acc, false, false);
                }
                const int mrow = mt * 16 + ((lane >> 4) << 3);
                const int colg = t2base + nt * 16 + nRow;
                float* op = out + ((size_t)b * QNA_HID + mrow) * QNA_T2 + colg;
#pragma unroll
                for (int j = 0; j < 8; ++j)
                    op[(size_t)j * QNA_T2] = acc[j] + pjb_l[mrow + j];
            }
        }
    }
}

// =====================================================================
extern "C" void kernel_launch(void* const* d_in, const int* in_sizes, int n_in,
                              void* d_out, int out_size, void* d_ws, size_t ws_size,
                              hipStream_t stream) {
    const float* x      = (const float*)d_in[0];
    const float* temb   = (const float*)d_in[1];
    const float* ln_w   = (const float*)d_in[2];
    const float* ln_b   = (const float*)d_in[3];
    const float* ss_w   = (const float*)d_in[4];
    const float* ss_b   = (const float*)d_in[5];
    const float* kv_w   = (const float*)d_in[6];
    const float* kv_b   = (const float*)d_in[7];
    const float* q      = (const float*)d_in[8];
    const float* qt_w   = (const float*)d_in[9];
    const float* qt_b   = (const float*)d_in[10];
    const float* rpe    = (const float*)d_in[11];
    const float* proj_w = (const float*)d_in[12];
    const float* proj_b = (const float*)d_in[13];

    char* ws = (char*)d_ws;
    __bf16* kvpk = (__bf16*)(ws + WS_KVPK);
    __bf16* pjpk = (__bf16*)(ws + WS_PJPK);
    float*  cond = (float*)(ws + WS_COND);
    float*  qb   = (float*)(ws + WS_QB);

    // pack weights into WMMA A-fragment bf16 layout ((256+128) frags * 32 lanes)
    qna_pack_w<<<48, 256, 0, stream>>>(kv_w, proj_w, kvpk, pjpk);
    // per-batch FiLM cond + timestep query
    qna_cond<<<QNA_B, 256, 0, stream>>>(temb, ss_w, ss_b, qt_w, qt_b, q, cond, qb);

    // main fused kernel: 16 batches * 128 tiles
    (void)hipFuncSetAttribute((const void*)qna_main,
                              hipFuncAttributeMaxDynamicSharedMemorySize,
                              SMEM_BYTES);
    qna_main<<<QNA_B * 128, 256, SMEM_BYTES, stream>>>(
        x, ln_w, ln_b, kv_b, rpe, proj_b, cond, qb, kvpk, pjpk, (float*)d_out);
}
